// SelfAttention_65249143161604
// MI455X (gfx1250) — compile-verified, matrix operand order checked by reading, not run on hardware
//
#include <hip/hip_runtime.h>

// ---------- types ----------
typedef __attribute__((ext_vector_type(16))) __bf16 v16bf;
typedef __attribute__((ext_vector_type(8)))  float  v8f;
typedef __attribute__((ext_vector_type(8)))  unsigned short ushortx8;
typedef __attribute__((ext_vector_type(4)))  float  f32x4;

union FragBF {
    v16bf v;
    ushortx8 h[2];
};

// ---------- constants ----------
#define M_DIM 16384   // B*S = 8*2048
#define N_DIM 2048    // Dm
#define K_DIM 2048    // Dm
#define TM 128
#define TN 128
#define TK 64
#define LDS_STRIDE 72         // 64 + 8 pad shorts -> 144B row stride, bank-spread
#define NT (K_DIM / TK)       // 32 K-tiles
#define BUF_BYTES (TM * LDS_STRIDE * 2)

// ---------- CDNA5 async global->LDS copy (512B per instruction per wave) ----------
// Each lane copies 16B from its global address to its own LDS byte address.
#define ASYNC_LDS_COPY_B128(ldsaddr, gptr)                                      \
    asm volatile("global_load_async_to_lds_b128 %0, %1, off"                    \
                 :: "v"(ldsaddr), "v"((unsigned long long)(uintptr_t)(gptr))    \
                 : "memory")

#define WAIT_ASYNCCNT(n) asm volatile("s_wait_asynccnt %0" :: "i"(n) : "memory")

// ---------- fp32 -> bf16 (round-to-nearest-even) ----------
__device__ __forceinline__ unsigned short f32_to_bf16(float f) {
    unsigned int u = __float_as_uint(f);
    unsigned int r = u + 0x7FFFu + ((u >> 16) & 1u);
    return (unsigned short)(r >> 16);
}

// linear convert (activations)
__global__ __launch_bounds__(256) void cvt_f32_bf16(const float* __restrict__ src,
                                                    unsigned short* __restrict__ dst,
                                                    int n4) {
    int i = blockIdx.x * blockDim.x + threadIdx.x;
    if (i < n4) {
        f32x4 v = *(const f32x4*)(src + i * 4);
        unsigned long long packed =
            (unsigned long long)f32_to_bf16(v.x) |
            ((unsigned long long)f32_to_bf16(v.y) << 16) |
            ((unsigned long long)f32_to_bf16(v.z) << 32) |
            ((unsigned long long)f32_to_bf16(v.w) << 48);
        *(unsigned long long*)(dst + i * 4) = packed;
    }
}

// convert + transpose (weights): src is K x N fp32, dst is N x K bf16
__global__ __launch_bounds__(256) void cvt_transpose_f32_bf16(const float* __restrict__ src,
                                                              unsigned short* __restrict__ dst) {
    __shared__ float tile[32][33];
    const int bn = blockIdx.x * 32;  // n base
    const int bk = blockIdx.y * 32;  // k base
    const int tx = threadIdx.x & 31;
    const int ty = threadIdx.x >> 5;  // 0..7
#pragma unroll
    for (int i = 0; i < 32; i += 8)
        tile[ty + i][tx] = src[(size_t)(bk + ty + i) * N_DIM + bn + tx];
    __syncthreads();
#pragma unroll
    for (int i = 0; i < 32; i += 8)
        dst[(size_t)(bn + ty + i) * K_DIM + bk + tx] = f32_to_bf16(tile[tx][ty + i]);
}

// ---------- GEMM (bf16 WMMA) + optional fused RoPE epilogue ----------
// C[M,N] = A[M,K] * Bt[N,K]^T ; block tile 128x128, K-step 64, double-buffered
// async-to-LDS staging. 8 waves: waveM 0..3 (32 rows), waveN 0..1 (64 cols);
// each wave: 2x4 grid of 16x16 f32 accumulators.
__global__ __launch_bounds__(256) void qkv_gemm_rope(
    const unsigned short* __restrict__ A,    // M x K bf16 (row-major)
    const unsigned short* __restrict__ Bt,   // N x K bf16 (pre-transposed weight)
    float* __restrict__ C,                   // M x N f32
    int do_rope)
{
    __shared__ unsigned short As[2][TM][LDS_STRIDE];
    __shared__ unsigned short Bs[2][TN][LDS_STRIDE];

    const int tid   = threadIdx.x;
    const int lane  = tid & 31;
    const int wv    = tid >> 5;
    const int waveM = wv & 3;
    const int waveN = wv >> 2;
    const int l16   = lane & 15;
    const int hi    = lane >> 4;   // 0: K 0-7/16-23, 1: K 8-15/24-31 (ISA 16-bit layout)

    const int blockN = blockIdx.x * TN;
    const int blockM = blockIdx.y * TM;

    // LDS byte addresses (low 32 bits of a flat LDS pointer = wave-relative LDS offset)
    const unsigned asBase = (unsigned)(uintptr_t)&As[0][0][0];
    const unsigned bsBase = (unsigned)(uintptr_t)&Bs[0][0][0];

    v8f acc[2][4];
#pragma unroll
    for (int i = 0; i < 2; ++i)
#pragma unroll
        for (int j = 0; j < 4; ++j)
            acc[i][j] = v8f{0.f, 0.f, 0.f, 0.f, 0.f, 0.f, 0.f, 0.f};

    // Issue one 128x64 bf16 tile each for A and B into LDS buffer `buf`
    // (1024 16B chunks per matrix; 32 lanes/instr -> 4 async instrs per wave per matrix).
    auto issue_tile = [&](int kb, int buf) {
        const unsigned bufOff = (unsigned)buf * BUF_BYTES;
#pragma unroll
        for (int i = 0; i < 4; ++i) {
            const int chunk = ((wv * 4 + i) << 5) | lane;   // 0..1023
            const int r  = chunk >> 3;                       // row 0..127
            const int c8 = (chunk & 7) << 3;                 // k offset in shorts
            const unsigned loff = bufOff + (unsigned)(r * LDS_STRIDE + c8) * 2u;
            ASYNC_LDS_COPY_B128(asBase + loff, A  + (size_t)(blockM + r) * K_DIM + kb + c8);
            ASYNC_LDS_COPY_B128(bsBase + loff, Bt + (size_t)(blockN + r) * K_DIM + kb + c8);
        }
    };

    auto compute_tile = [&](int buf) {
#pragma unroll
        for (int ks = 0; ks < TK; ks += 32) {
            const int k0 = ks + hi * 8;
            FragBF a[2];
#pragma unroll
            for (int i = 0; i < 2; ++i) {
                const int row = waveM * 32 + i * 16 + l16;
                a[i].h[0] = *(const ushortx8*)&As[buf][row][k0];
                a[i].h[1] = *(const ushortx8*)&As[buf][row][k0 + 16];
            }
            FragBF b[4];
#pragma unroll
            for (int j = 0; j < 4; ++j) {
                const int n = waveN * 64 + j * 16 + l16;
                b[j].h[0] = *(const ushortx8*)&Bs[buf][n][k0];
                b[j].h[1] = *(const ushortx8*)&Bs[buf][n][k0 + 16];
            }
#pragma unroll
            for (int i = 0; i < 2; ++i)
#pragma unroll
                for (int j = 0; j < 4; ++j)
                    acc[i][j] = __builtin_amdgcn_wmma_f32_16x16x32_bf16(
                        false, a[i].v, false, b[j].v,
                        (short)0, acc[i][j], false, false);
        }
    };

    // ---- software pipeline: prefetch t+1 while computing t ----
    issue_tile(0, 0);
    int buf = 0;
    for (int t = 0; t < NT - 1; ++t) {
        issue_tile((t + 1) * TK, buf ^ 1);
        if (t + 2 < NT) {  // L2 prefetch two tiles ahead
            __builtin_prefetch(A  + (size_t)(blockM + (tid & 127)) * K_DIM + (t + 2) * TK, 0, 0);
            __builtin_prefetch(Bt + (size_t)(blockN + (tid & 127)) * K_DIM + (t + 2) * TK, 0, 0);
        }
        WAIT_ASYNCCNT(8);   // in-order: tile t's 8 ops have landed (t+1's 8 still in flight)
        __syncthreads();
        compute_tile(buf);
        __syncthreads();    // everyone done reading `buf` before it is refilled next iter
        buf ^= 1;
    }
    WAIT_ASYNCCNT(0);
    __syncthreads();
    compute_tile(buf);

    // ---- epilogue: optional RoPE (pos = head index), then store ----
#pragma unroll
    for (int j = 0; j < 4; ++j) {
        const int col = blockN + waveN * 64 + j * 16 + l16;
        float s = 0.f, c = 1.f;
        if (do_rope) {
            const int h  = col >> 7;          // head index = "position"
            const int ii = (col & 127) >> 1;  // frequency index
            // theta_i = 10000^(-2i/128); ang = h * theta_i
            float ang = (float)h * __expf(-(float)(2 * ii) * (9.210340372f / 128.0f));
            __sincosf(ang, &s, &c);
        }
#pragma unroll
        for (int i = 0; i < 2; ++i) {
            const int rowBase = blockM + waveM * 32 + i * 16 + hi * 8;
#pragma unroll
            for (int r = 0; r < 8; ++r) {
                float v = acc[i][j][r];
                float out = v;
                if (do_rope) {
                    float other = __shfl_xor(v, 1, 32);  // even<->odd column pair
                    out = ((col & 1) == 0) ? (v * c - other * s)   // re
                                           : (other * s + v * c);  // ro
                }
                C[(size_t)(rowBase + r) * N_DIM + col] = out;
            }
        }
    }
}

// ---------- launcher ----------
extern "C" void kernel_launch(void* const* d_in, const int* in_sizes, int n_in,
                              void* d_out, int out_size, void* d_ws, size_t ws_size,
                              hipStream_t stream) {
    const float* x  = (const float*)d_in[0];
    const float* Wq = (const float*)d_in[1];
    const float* Wk = (const float*)d_in[2];
    const float* Wv = (const float*)d_in[3];
    // d_in[4] = st_pos (unused by the reference computation)

    const size_t xElems = (size_t)M_DIM * K_DIM;   // 33,554,432
    const size_t wElems = (size_t)K_DIM * N_DIM;   //  4,194,304

    unsigned short* xb  = (unsigned short*)d_ws;
    unsigned short* wqt = xb  + xElems;            // transposed bf16 weights (N x K)
    unsigned short* wkt = wqt + wElems;
    unsigned short* wvt = wkt + wElems;

    // fp32 -> bf16 staging
    {
        int n4 = (int)(xElems / 4);
        cvt_f32_bf16<<<(n4 + 255) / 256, 256, 0, stream>>>(x, xb, n4);
        dim3 tg(N_DIM / 32, K_DIM / 32);  // (64, 64)
        cvt_transpose_f32_bf16<<<tg, 256, 0, stream>>>(Wq, wqt);
        cvt_transpose_f32_bf16<<<tg, 256, 0, stream>>>(Wk, wkt);
        cvt_transpose_f32_bf16<<<tg, 256, 0, stream>>>(Wv, wvt);
    }

    float* out = (float*)d_out;
    dim3 grid(N_DIM / TN, M_DIM / TM);  // (16, 128)
    dim3 block(256);

    qkv_gemm_rope<<<grid, block, 0, stream>>>(xb, wqt, out,              1); // Q + rope
    qkv_gemm_rope<<<grid, block, 0, stream>>>(xb, wkt, out + xElems,     1); // K + rope
    qkv_gemm_rope<<<grid, block, 0, stream>>>(xb, wvt, out + 2 * xElems, 0); // V
}
